// res_atom_exchange_14482629722557
// MI455X (gfx1250) — compile-verified, matrix operand order checked by reading, not run on hardware
//
#include <hip/hip_runtime.h>
#include <stdint.h>

// ---------------------------------------------------------------------------
// res_atom_exchange for MI455X (gfx1250): bf16 WMMA pipeline
//   scatter -> QKV proj (WMMA GEMM, double-buffered) -> flash attention (WMMA,
//   transposed-V fragments, prefetch) -> out proj -> second attention pass ->
//   LayerNorm + residual + gather
// ---------------------------------------------------------------------------

typedef __bf16 bf16;
typedef __attribute__((ext_vector_type(16))) __bf16 v16bf;
typedef __attribute__((ext_vector_type(8)))  float  v8f;

#define BATCH 16
#define DMODEL 512
#define NHEAD 8
#define DHEAD 64
#define MAQ 2048
#define MRQ 256

struct BInfo  { int cnt[BATCH]; int start[BATCH]; };
struct BCounts{ int c[BATCH]; };

// ---- scalar f32 -> bf16 (RNE) ---------------------------------------------
static __device__ __forceinline__ bf16 f2b(float f) {
  union { float f; uint32_t u; } x; x.f = f;
  uint32_t r = x.u + 0x7FFFu + ((x.u >> 16) & 1u);
  uint16_t h = (uint16_t)(r >> 16);
  return __builtin_bit_cast(__bf16, h);
}

union FragU { v16bf v; uint4 q[2]; };

// Load a 16x32 bf16 WMMA fragment (A-layout; also valid as B when memory is
// [N,K] row-major): lane (l&15) -> row, k-chunks [koff,koff+8) and
// [16+koff,16+koff+8), koff = (l>>4)*8. Two 16B vector loads per lane.
static __device__ __forceinline__ v16bf load_frag_rm(const bf16* base, int ld, int lane) {
  int r = lane & 15, koff = (lane >> 4) * 8;
  const bf16* p = base + (size_t)r * ld + koff;
  FragU f;
  f.q[0] = *(const uint4*)(p);
  f.q[1] = *(const uint4*)(p + 16);
  return f.v;
}

static __device__ __forceinline__ v8f wmma_bf16(v16bf a, v16bf b, v8f c) {
  return __builtin_amdgcn_wmma_f32_16x16x32_bf16(false, a, false, b, (short)0, c, false, false);
}

// ---------------------------------------------------------------------------
// f32 -> bf16 conversion (weights)
__global__ void cvt_bf16_kernel(const float* __restrict__ src, bf16* __restrict__ dst, int n) {
  int i = blockIdx.x * blockDim.x + threadIdx.x;
  if (i < n) dst[i] = f2b(src[i]);
}

// flat [N,D] f32 -> dense [B,Mq,D] bf16, zero padded
__global__ void scatter_dense_kernel(const float* __restrict__ flat, bf16* __restrict__ dense,
                                     int Mq, BInfo info) {
  size_t idx = (size_t)blockIdx.x * blockDim.x + threadIdx.x;
  int d = (int)(idx & (DMODEL - 1));
  size_t rm = idx >> 9;
  int m = (int)(rm % Mq);
  int b = (int)(rm / Mq);
  float v = 0.0f;
  if (m < info.cnt[b]) v = flat[(size_t)(info.start[b] + m) * DMODEL + d];
  dense[idx] = f2b(v);
}

// ---------------------------------------------------------------------------
// C[M,N] = X[M,512] @ W[N,512]^T + bias   (bf16 in, f32 accum)
// block = 256 thr = 8 waves; wave tile 16(M)x64(N); block tile 32x256.
// grid = (M/32, N/256). Register double-buffered K loop (load k+32 while the
// matrix pipe runs k) so waits need not drain LOADcnt to zero.
// vtMk > 0: store bf16 output in transposed per-head V layout [B,H,64,vtMk].
__global__ __launch_bounds__(256) void gemm_xwT_kernel(
    const bf16* __restrict__ X, const bf16* __restrict__ W,
    const float* __restrict__ bias, bf16* __restrict__ outB,
    float* __restrict__ outF, int M, int N, int vtMk) {
  const int K = DMODEL;
  int lane = threadIdx.x & 31, wave = threadIdx.x >> 5;
  int wm = wave & 1, wn = wave >> 1;
  int m0 = blockIdx.x * 32 + wm * 16;
  int n0 = blockIdx.y * 256 + wn * 64;

  const bf16* Xp = X + (size_t)m0 * K;
  const bf16* Wp = W + (size_t)n0 * K;

  v8f acc[4] = {};
  v16bf a = load_frag_rm(Xp, K, lane);
  v16bf w[4];
#pragma unroll
  for (int nt = 0; nt < 4; ++nt)
    w[nt] = load_frag_rm(Wp + (size_t)(nt * 16) * K, K, lane);

#pragma unroll
  for (int k = 0; k < K; k += 32) {
    v16bf an, wn0, wn1, wn2, wn3;
    const bool more = (k + 32 < K);
    if (more) {
      an  = load_frag_rm(Xp + k + 32, K, lane);
      wn0 = load_frag_rm(Wp + (size_t)(0 * 16) * K + k + 32, K, lane);
      wn1 = load_frag_rm(Wp + (size_t)(1 * 16) * K + k + 32, K, lane);
      wn2 = load_frag_rm(Wp + (size_t)(2 * 16) * K + k + 32, K, lane);
      wn3 = load_frag_rm(Wp + (size_t)(3 * 16) * K + k + 32, K, lane);
    }
    acc[0] = wmma_bf16(a, w[0], acc[0]);
    acc[1] = wmma_bf16(a, w[1], acc[1]);
    acc[2] = wmma_bf16(a, w[2], acc[2]);
    acc[3] = wmma_bf16(a, w[3], acc[3]);
    if (more) { a = an; w[0] = wn0; w[1] = wn1; w[2] = wn2; w[3] = wn3; }
  }

  int rl = lane & 15, hi = lane >> 4;
#pragma unroll
  for (int nt = 0; nt < 4; ++nt) {
    int n = n0 + nt * 16 + rl;
    float bv = bias[n];
#pragma unroll
    for (int j = 0; j < 8; ++j) {
      int m = m0 + hi * 8 + j;
      float v = acc[nt][j] + bv;
      if (vtMk > 0) {
        int bb = m / vtMk, mk = m - bb * vtMk;
        int h = n >> 6, nd = n & 63;
        size_t o = (((size_t)bb * NHEAD + h) * DHEAD + nd) * (size_t)vtMk + mk;
        outB[o] = f2b(v);
      } else {
        size_t o = (size_t)m * N + n;
        outB[o] = f2b(v);
        if (outF) outF[o] = v;
      }
    }
  }
}

// ---------------------------------------------------------------------------
// Flash attention: grid (Mq/16, B), block 256 = 8 waves, wave w = head w.
// Q,K dense bf16 [B,M,512]; V transposed bf16 [B,H,64,Mk] so the P@V B-frags
// are plain 16B vector loads. Online softmax over key tiles of 32; only
// ceil(cnt/32) tiles visited (padded keys skipped). Next tile prefetched.
__global__ __launch_bounds__(256) void flash_attn_kernel(
    const bf16* __restrict__ Q, const bf16* __restrict__ Kd,
    const bf16* __restrict__ Vt, bf16* __restrict__ CTX,
    int Mq, int Mk, BCounts kc) {
  __shared__ bf16 plds[NHEAD][16 * 32];

  int lane = threadIdx.x & 31, h = threadIdx.x >> 5;
  int rl = lane & 15, hi = lane >> 4;
  int b = blockIdx.y, m0 = blockIdx.x * 16;
  int cnt = kc.c[b];

  const bf16* Qb = Q  + ((size_t)b * Mq + m0) * DMODEL + h * DHEAD;
  const bf16* Kb = Kd + (size_t)b * Mk * DMODEL + h * DHEAD;
  const bf16* Vb = Vt + ((size_t)b * NHEAD + h) * DHEAD * (size_t)Mk;

  v16bf qf0 = load_frag_rm(Qb, DMODEL, lane);
  v16bf qf1 = load_frag_rm(Qb + 32, DMODEL, lane);

  v8f ctx[4] = {};
  float mrow[8], lrow[8];
#pragma unroll
  for (int j = 0; j < 8; ++j) { mrow[j] = -1e30f; lrow[j] = 0.0f; }

  for (int kb = 0; kb < cnt; kb += 32) {
    // ---- batch all K fragments for this 32-key tile into one load clause
    v16bf kf[4];
    {
      const bf16* Kt0 = Kb + (size_t)kb * DMODEL;
      const bf16* Kt1 = Kb + (size_t)(kb + 16) * DMODEL;
      kf[0] = load_frag_rm(Kt0,      DMODEL, lane);
      kf[1] = load_frag_rm(Kt0 + 32, DMODEL, lane);
      kf[2] = load_frag_rm(Kt1,      DMODEL, lane);
      kf[3] = load_frag_rm(Kt1 + 32, DMODEL, lane);
    }
    if (kb + 32 < cnt) {   // warm next key tile while softmax serializes
      __builtin_prefetch(Kb + (size_t)(kb + 32) * DMODEL, 0, 0);
      __builtin_prefetch(Vb + kb + 32, 0, 0);
    }

    // ---- scores for 32 keys (two 16-key subtiles), K-dim = 64 = 2 wmma each
    float s[2][8];
#pragma unroll
    for (int sub = 0; sub < 2; ++sub) {
      v8f sc = {};
      sc = wmma_bf16(qf0, kf[sub * 2],     sc);
      sc = wmma_bf16(qf1, kf[sub * 2 + 1], sc);
      int key = kb + sub * 16 + rl;
      bool valid = key < cnt;
#pragma unroll
      for (int j = 0; j < 8; ++j)
        s[sub][j] = valid ? sc[j] * 0.125f : -1e30f;
    }

    // ---- online softmax update (row stats live per (hi,j) == C-layout row)
    float scl[8];
#pragma unroll
    for (int j = 0; j < 8; ++j) {
      float t = fmaxf(s[0][j], s[1][j]);
#pragma unroll
      for (int off = 8; off >= 1; off >>= 1)
        t = fmaxf(t, __shfl_xor(t, off, 32));
      float mn = fmaxf(mrow[j], t);
      float sc = __expf(mrow[j] - mn);
      mrow[j] = mn;
      float p0 = __expf(s[0][j] - mn);
      float p1 = __expf(s[1][j] - mn);
      int row = hi * 8 + j;
      plds[h][row * 32 + rl]      = f2b(p0);
      plds[h][row * 32 + 16 + rl] = f2b(p1);
      float rs = p0 + p1;
#pragma unroll
      for (int off = 8; off >= 1; off >>= 1)
        rs += __shfl_xor(rs, off, 32);
      lrow[j] = lrow[j] * sc + rs;
      scl[j] = sc;
    }
#pragma unroll
    for (int nt = 0; nt < 4; ++nt)
#pragma unroll
      for (int j = 0; j < 8; ++j)
        ctx[nt][j] *= scl[j];

    // ---- P (C-layout) -> A-layout via LDS, then ctx += P @ V
    v16bf pf;
    {
      const bf16* p = &plds[h][rl * 32 + hi * 8];
#pragma unroll
      for (int i = 0; i < 8; ++i) { pf[i] = p[i]; pf[i + 8] = p[16 + i]; }
    }
#pragma unroll
    for (int nt = 0; nt < 4; ++nt) {
      // B(k,n) = V[key=k][dh=n] = Vt[dh-row][key-col], k contiguous -> b128s
      v16bf vf = load_frag_rm(Vb + (size_t)(nt * 16) * Mk + kb, Mk, lane);
      ctx[nt] = wmma_bf16(pf, vf, ctx[nt]);
    }
  }

  // ---- normalize and store context (bf16, row-major dense)
  bf16* Cb = CTX + ((size_t)b * Mq + m0) * DMODEL + h * DHEAD;
#pragma unroll
  for (int j = 0; j < 8; ++j) {
    float inv = 1.0f / lrow[j];
    int row = hi * 8 + j;
#pragma unroll
    for (int nt = 0; nt < 4; ++nt)
      Cb[(size_t)row * DMODEL + nt * 16 + rl] = f2b(ctx[nt][j] * inv);
  }
}

// ---------------------------------------------------------------------------
// LayerNorm + gather-back + residual. One block per flat row.
__global__ __launch_bounds__(256) void ln_gather_kernel(
    const float* __restrict__ upd, const float* __restrict__ g,
    const float* __restrict__ bb, const float* __restrict__ flat_in,
    const int* __restrict__ batch, float* __restrict__ out,
    int Mq, BInfo info) {
  __shared__ float red[2][8];
  int i = blockIdx.x;
  int b = batch[i];
  int pos = i - info.start[b];
  const float* row = upd + ((size_t)b * Mq + pos) * DMODEL;
  int t = threadIdx.x;
  float v0 = row[2 * t], v1 = row[2 * t + 1];
  float s = v0 + v1, sq = v0 * v0 + v1 * v1;
#pragma unroll
  for (int off = 16; off >= 1; off >>= 1) {
    s  += __shfl_xor(s, off, 32);
    sq += __shfl_xor(sq, off, 32);
  }
  int lane = t & 31, w = t >> 5;
  if (lane == 0) { red[0][w] = s; red[1][w] = sq; }
  __syncthreads();
  if (t == 0) {
    float ts = 0.f, tq = 0.f;
#pragma unroll
    for (int k = 0; k < 8; ++k) { ts += red[0][k]; tq += red[1][k]; }
    red[0][0] = ts; red[1][0] = tq;
  }
  __syncthreads();
  float mean = red[0][0] * (1.0f / DMODEL);
  float var  = red[1][0] * (1.0f / DMODEL) - mean * mean;
  float rstd = rsqrtf(var + 1e-5f);
  size_t o = (size_t)i * DMODEL + 2 * t;
  out[o]     = (v0 - mean) * rstd * g[2 * t]     + bb[2 * t]     + flat_in[o];
  out[o + 1] = (v1 - mean) * rstd * g[2 * t + 1] + bb[2 * t + 1] + flat_in[o + 1];
}

// ---------------------------------------------------------------------------
extern "C" void kernel_launch(void* const* d_in, const int* in_sizes, int n_in,
                              void* d_out, int out_size, void* d_ws, size_t ws_size,
                              hipStream_t stream) {
  const float* atom_node = (const float*)d_in[0];
  const float* res_node  = (const float*)d_in[1];
  const float* r2a_in_w  = (const float*)d_in[2];
  const float* r2a_in_b  = (const float*)d_in[3];
  const float* r2a_out_w = (const float*)d_in[4];
  const float* r2a_out_b = (const float*)d_in[5];
  const float* a2r_in_w  = (const float*)d_in[6];
  const float* a2r_in_b  = (const float*)d_in[7];
  const float* a2r_out_w = (const float*)d_in[8];
  const float* a2r_out_b = (const float*)d_in[9];
  const float* atom_ln_g = (const float*)d_in[10];
  const float* atom_ln_b = (const float*)d_in[11];
  const float* res_ln_g  = (const float*)d_in[12];
  const float* res_ln_b  = (const float*)d_in[13];
  const int*   atom_batch = (const int*)d_in[14];
  const int*   res_batch  = (const int*)d_in[15];

  int Na = in_sizes[0] / DMODEL;
  int Nr = in_sizes[1] / DMODEL;

  // deterministic per-batch counts (reference's _counts())
  BInfo ai, ri; BCounts ac, rc;
  int sa = 0, sr = 0;
  for (int i = 0; i < BATCH; ++i) {
    int ca = 1024 + (i * 97) % 1024;
    int cr = 128  + (i * 37) % 128;
    ai.cnt[i] = ca; ai.start[i] = sa; sa += ca;
    ri.cnt[i] = cr; ri.start[i] = sr; sr += cr;
    ac.c[i] = ca; rc.c[i] = cr;
  }

  // ---- workspace layout (256B aligned slices, aggressive reuse) ----
  char* ws = (char*)d_ws;
  size_t off = 0;
  auto wsalloc = [&](size_t bytes) -> void* {
    void* p = ws + off;
    off = (off + bytes + 255) & ~(size_t)255;
    return p;
  };
  const size_t ADN = (size_t)BATCH * MAQ * DMODEL;   // dense atom elems
  const size_t RDN = (size_t)BATCH * MRQ * DMODEL;   // dense res elems

  bf16*  A_DENSE = (bf16*)wsalloc(ADN * 2);          // later: AUPD bf16
  bf16*  R_DENSE = (bf16*)wsalloc(RDN * 2);
  bf16*  WQKV1   = (bf16*)wsalloc((size_t)3 * DMODEL * DMODEL * 2);
  bf16*  WO1     = (bf16*)wsalloc((size_t)DMODEL * DMODEL * 2);
  bf16*  WQKV2   = (bf16*)wsalloc((size_t)3 * DMODEL * DMODEL * 2);
  bf16*  WO2     = (bf16*)wsalloc((size_t)DMODEL * DMODEL * 2);
  bf16*  QA      = (bf16*)wsalloc(ADN * 2);          // later: KA
  bf16*  KR      = (bf16*)wsalloc(RDN * 2);          // later: QR, then dump
  bf16*  VRt     = (bf16*)wsalloc(RDN * 2);          // later: CTXR
  bf16*  CTXA    = (bf16*)wsalloc(ADN * 2);          // later: VAt
  float* AUPD    = (float*)wsalloc(ADN * 4);
  float* RUPD    = (float*)wsalloc(RDN * 4);
  (void)ws_size; (void)n_in; (void)out_size;

  const int WSZ = DMODEL * DMODEL;                   // 262144

  // ---- weights to bf16 ----
  cvt_bf16_kernel<<<(3 * WSZ) / 256, 256, 0, stream>>>(r2a_in_w,  WQKV1, 3 * WSZ);
  cvt_bf16_kernel<<<WSZ / 256,       256, 0, stream>>>(r2a_out_w, WO1,   WSZ);
  cvt_bf16_kernel<<<(3 * WSZ) / 256, 256, 0, stream>>>(a2r_in_w,  WQKV2, 3 * WSZ);
  cvt_bf16_kernel<<<WSZ / 256,       256, 0, stream>>>(a2r_out_w, WO2,   WSZ);

  // ---- flat -> dense (zero padded) ----
  scatter_dense_kernel<<<(unsigned)(ADN / 256), 256, 0, stream>>>(atom_node, A_DENSE, MAQ, ai);
  scatter_dense_kernel<<<(unsigned)(RDN / 256), 256, 0, stream>>>(res_node,  R_DENSE, MRQ, ri);

  const int MA_ROWS = BATCH * MAQ;   // 32768
  const int MR_ROWS = BATCH * MRQ;   // 4096

  // ---- phase A: r2a attention (q = atoms, k/v = res) ----
  gemm_xwT_kernel<<<dim3(MA_ROWS / 32, 2), 256, 0, stream>>>(
      A_DENSE, WQKV1,            r2a_in_b,              QA,  nullptr, MA_ROWS, DMODEL, 0);
  gemm_xwT_kernel<<<dim3(MR_ROWS / 32, 2), 256, 0, stream>>>(
      R_DENSE, WQKV1 + WSZ,      r2a_in_b + DMODEL,     KR,  nullptr, MR_ROWS, DMODEL, 0);
  gemm_xwT_kernel<<<dim3(MR_ROWS / 32, 2), 256, 0, stream>>>(
      R_DENSE, WQKV1 + 2 * WSZ,  r2a_in_b + 2 * DMODEL, VRt, nullptr, MR_ROWS, DMODEL, MRQ);

  flash_attn_kernel<<<dim3(MAQ / 16, BATCH), 256, 0, stream>>>(
      QA, KR, VRt, CTXA, MAQ, MRQ, rc);

  // out projection: AUPD (f32 for LN) + A_DENSE reused as AUPD bf16 (K/V input)
  gemm_xwT_kernel<<<dim3(MA_ROWS / 32, 2), 256, 0, stream>>>(
      CTXA, WO1, r2a_out_b, A_DENSE, AUPD, MA_ROWS, DMODEL, 0);

  // ---- phase B: a2r attention (q = res, k/v = atom_upd) ----
  gemm_xwT_kernel<<<dim3(MR_ROWS / 32, 2), 256, 0, stream>>>(
      R_DENSE, WQKV2,           a2r_in_b,              KR /*QR*/,   nullptr, MR_ROWS, DMODEL, 0);
  gemm_xwT_kernel<<<dim3(MA_ROWS / 32, 2), 256, 0, stream>>>(
      A_DENSE, WQKV2 + WSZ,     a2r_in_b + DMODEL,     QA /*KA*/,   nullptr, MA_ROWS, DMODEL, 0);
  gemm_xwT_kernel<<<dim3(MA_ROWS / 32, 2), 256, 0, stream>>>(
      A_DENSE, WQKV2 + 2 * WSZ, a2r_in_b + 2 * DMODEL, CTXA /*VAt*/, nullptr, MA_ROWS, DMODEL, MAQ);

  flash_attn_kernel<<<dim3(MRQ / 16, BATCH), 256, 0, stream>>>(
      KR /*QR*/, QA /*KA*/, CTXA /*VAt*/, VRt /*CTXR*/, MRQ, MAQ, ac);

  gemm_xwT_kernel<<<dim3(MR_ROWS / 32, 2), 256, 0, stream>>>(
      VRt /*CTXR*/, WO2, a2r_out_b, KR /*dump*/, RUPD, MR_ROWS, DMODEL, 0);

  // ---- LayerNorm + residual + gather to flat output ----
  float* out_atoms = (float*)d_out;
  float* out_res   = out_atoms + (size_t)Na * DMODEL;
  ln_gather_kernel<<<Na, 256, 0, stream>>>(AUPD, atom_ln_g, atom_ln_b,
                                           atom_node, atom_batch, out_atoms, MAQ, ai);
  ln_gather_kernel<<<Nr, 256, 0, stream>>>(RUPD, res_ln_g, res_ln_b,
                                           res_node, res_batch, out_res, MRQ, ri);
}
